// DepthVolume1D_6468220748329
// MI455X (gfx1250) — matrix operand order repaired and checked by reading, not compiled, not run on hardware
//
#include <hip/hip_runtime.h>
#include <hip/hip_bf16.h>

typedef __attribute__((ext_vector_type(16))) _Float16 v16h;
typedef __attribute__((ext_vector_type(8)))  _Float16 v8h;
typedef __attribute__((ext_vector_type(8)))  float    v8f;

#define DEV __device__ __forceinline__

static constexpr int DNUM = 16, NVIEW = 4, NCH = 32, IH = 128, IW = 160;
static constexpr int HWFULL = IH * IW;   // 20480
static constexpr int WCC = 40;           // 34 channels padded to 40 (x8)

// ---------------------------------------------------------------- small math
DEV void inv3x3(const float* m, float* o) {
  float a=m[0],b=m[1],c=m[2],d=m[3],e=m[4],f=m[5],g=m[6],h=m[7],i=m[8];
  float A =  (e*i - f*h), B = -(d*i - f*g), C =  (d*h - e*g);
  float det = a*A + b*B + c*C;
  float r = 1.0f / det;
  o[0]=A*r;            o[1]=-(b*i-c*h)*r;  o[2]=(b*f-c*e)*r;
  o[3]=B*r;            o[4]=(a*i-c*g)*r;   o[5]=-(a*f-c*d)*r;
  o[6]=C*r;            o[7]=-(a*h-b*g)*r;  o[8]=(a*e-b*d)*r;
}

// Fold projection chain into per-view affine map:  q = depth*(A_v @ [x,y,1]) + b_v
__global__ void setup_mats_kernel(const float* __restrict__ dstK, const float* __restrict__ dstE,
                                  const float* __restrict__ srcK, const float* __restrict__ srcE,
                                  float* __restrict__ mats) {
  if (threadIdx.x != 0 || blockIdx.x != 0) return;
  float K3[9], iK[9];
  for (int i=0;i<3;++i) for (int j=0;j<3;++j) K3[i*3+j] = dstK[i*4+j];
  inv3x3(K3, iK);
  float R[9], t[3], iR[9];
  for (int i=0;i<3;++i){ for (int j=0;j<3;++j) R[i*3+j]=dstE[i*4+j]; t[i]=dstE[i*4+3]; }
  inv3x3(R, iR);
  float invE[16];
  for (int i=0;i<3;++i){
    for (int j=0;j<3;++j) invE[i*4+j]=iR[i*3+j];
    invE[i*4+3] = -(iR[i*3+0]*t[0] + iR[i*3+1]*t[1] + iR[i*3+2]*t[2]);
  }
  invE[12]=0.f; invE[13]=0.f; invE[14]=0.f; invE[15]=1.f;
  for (int v=0; v<NVIEW; ++v){
    const float* E  = srcE + v*16;
    const float* Kv = srcK + v*16;
    float M[12];
    for (int i=0;i<3;++i) for (int j=0;j<4;++j){
      float s=0.f; for (int k=0;k<4;++k) s += E[i*4+k]*invE[k*4+j];
      M[i*4+j]=s;
    }
    float P[12];
    for (int i=0;i<3;++i) for (int j=0;j<4;++j){
      float s=0.f; for (int k=0;k<3;++k) s += Kv[i*4+k]*M[k*4+j];
      P[i*4+j]=s;
    }
    for (int i=0;i<3;++i) for (int j=0;j<3;++j){
      float s=0.f; for (int k=0;k<3;++k) s += P[i*4+k]*iK[k*3+j];
      mats[v*12 + i*3 + j] = s;
    }
    for (int i=0;i<3;++i) mats[v*12 + 9 + i] = P[i*4+3];
  }
}

// --------------------------- warp (bilinear) -> NHWC f16 wc[n=(d*4+v)][p][c], c in [0,32)
__global__ __launch_bounds__(256) void warp_kernel(const float* __restrict__ feats,
                                                   const float* __restrict__ mats,
                                                   _Float16* __restrict__ wc) {
  int idx = blockIdx.x * blockDim.x + threadIdx.x;
  if (idx >= DNUM*NVIEW*HWFULL) return;
  int p = idx % HWFULL;
  int t = idx / HWFULL;
  int v = t % NVIEW;
  int d = t / NVIEW;
  int y = p / IW, x = p % IW;

  const float* A = mats + v*12;
  float depth = 0.5f + (float)d * (9.5f/15.0f);
  float fx = (float)x, fy = (float)y;
  float q0 = depth*(A[0]*fx + A[1]*fy + A[2]) + A[9];
  float q1 = depth*(A[3]*fx + A[4]*fy + A[5]) + A[10];
  float q2 = depth*(A[6]*fx + A[7]*fy + A[8]) + A[11];
  float iz = 1.0f / (q2 + 1e-6f);
  float u  = q0*iz, w = q1*iz;

  float x0f = floorf(u), y0f = floorf(w);
  float wx = u - x0f, wy = w - y0f;
  int x0 = (int)x0f, y0 = (int)y0f, x1 = x0+1, y1 = y0+1;
  bool vx0 = (x0>=0)&&(x0<IW), vx1 = (x1>=0)&&(x1<IW);
  bool vy0 = (y0>=0)&&(y0<IH), vy1 = (y1>=0)&&(y1<IH);
  float w00 = (1.f-wx)*(1.f-wy) * ((vx0&&vy0)?1.f:0.f);
  float w10 = wx*(1.f-wy)       * ((vx1&&vy0)?1.f:0.f);
  float w01 = (1.f-wx)*wy       * ((vx0&&vy1)?1.f:0.f);
  float w11 = wx*wy             * ((vx1&&vy1)?1.f:0.f);
  int xc0 = x0<0?0:(x0>IW-1?IW-1:x0), xc1 = x1<0?0:(x1>IW-1?IW-1:x1);
  int yc0 = y0<0?0:(y0>IH-1?IH-1:y0), yc1 = y1<0?0:(y1>IH-1?IH-1:y1);
  int o00 = yc0*IW+xc0, o10 = yc0*IW+xc1, o01 = yc1*IW+xc0, o11 = yc1*IW+xc1;

  const float* fb = feats + v*NCH*HWFULL;
  _Float16* ob = wc + ((t)*HWFULL + p) * WCC;   // t == n == d*4+v
  #pragma unroll
  for (int g=0; g<4; ++g){
    v8h o;
    #pragma unroll
    for (int j=0;j<8;++j){
      const float* f = fb + (g*8+j)*HWFULL;
      o[j] = (_Float16)(w00*f[o00] + w10*f[o10] + w01*f[o01] + w11*f[o11]);
    }
    *(v8h*)(ob + g*8) = o;
  }
}

// --------- cost channels (NHWC): ch32 = vcm, ch33 = view_cost[d,v], ch34..39 = 0 (pad)
__global__ __launch_bounds__(256) void cost_kernel(_Float16* __restrict__ wc) {
  int idx = blockIdx.x * blockDim.x + threadIdx.x;
  if (idx >= DNUM*HWFULL) return;
  int p = idx % HWFULL, d = idx / HWFULL;
  const int vstride = HWFULL * WCC;
  _Float16* b0 = wc + ((d*NVIEW)*HWFULL + p) * WCC;
  float s0=0.f,s1=0.f,s2=0.f,s3=0.f;
  #pragma unroll
  for (int g=0; g<4; ++g){
    v8h w0 = *(const v8h*)(b0 + g*8);
    v8h w1 = *(const v8h*)(b0 + g*8 + vstride);
    v8h w2 = *(const v8h*)(b0 + g*8 + 2*vstride);
    v8h w3 = *(const v8h*)(b0 + g*8 + 3*vstride);
    #pragma unroll
    for (int j=0;j<8;++j){
      float f0=(float)w0[j], f1=(float)w1[j], f2=(float)w2[j], f3=(float)w3[j];
      float a = 0.25f*(f0+f1+f2+f3);
      s0 += a*f0; s1 += a*f1; s2 += a*f2; s3 += a*f3;
    }
  }
  float sv[4] = {s0,s1,s2,s3};
  float vcm = 0.25f*(s0+s1+s2+s3);
  #pragma unroll
  for (int v=0; v<NVIEW; ++v){
    v8h o = {};
    o[0] = (_Float16)vcm;
    o[1] = (_Float16)sv[v];
    *(v8h*)(b0 + v*vstride + 32) = o;
  }
}

// ---- repack weights: OIHW f32 -> f16 [CoutPad16][KtotPad32], K ordered (tap, cin-padded)
__global__ __launch_bounds__(256) void repack_w_kernel(const float* __restrict__ w,
                                                       _Float16* __restrict__ o,
                                                       int Cout, int C, int Cpad,
                                                       int KK, int KtotPad, int total) {
  int idx = blockIdx.x * blockDim.x + threadIdx.x;
  if (idx >= total) return;
  int k  = idx % KtotPad;
  int co = idx / KtotPad;
  _Float16 val = (_Float16)0.f;
  int Ktot = KK * Cpad;
  if (co < Cout && k < Ktot) {
    int tap = k / Cpad;
    int cin = k - tap*Cpad;
    if (cin < C) val = (_Float16)w[(co*C + cin)*KK + tap];
  }
  o[idx] = val;
}

// ------------- fully-specialized implicit-GEMM conv via v_wmma_f32_16x16x32_f16 (NHWC)
// wave = 16 pixels (M) x NCT*16 out-ch (N); K = KH*KH*CIN in 32-chunks, (tap, cin) order.
// A tile loaded once per chunk and reused across NCT accumulators (NCT wmmas per chunk).
// Row index py is wave-uniform so vertical bounds checks scalarize.
template<int KH, int H, int W, int CIN, int C1, int COUT, int RELU, int UP, int NCT>
__global__ __launch_bounds__(256) void conv_wmma_kernel(
    const _Float16* __restrict__ inA, const _Float16* __restrict__ inB,
    const _Float16* __restrict__ wgtp, const float* __restrict__ bias,
    _Float16* __restrict__ out)
{
  constexpr int KK     = KH*KH;
  constexpr int PAD    = KH/2;
  constexpr int HW     = H*W;                 // multiple of 16 for all layers
  constexpr int TILESP = HW/16;
  constexpr int KTOT   = KK*CIN;
  constexpr int KTP    = (KTOT + 31) & ~31;
  constexpr int NCHUNK = (KTOT + 31)/32;
  constexpr int C2     = CIN - C1;
  constexpr int HS     = UP ? H/2 : H;
  constexpr int WS     = UP ? W/2 : W;

  const int wavei = (blockIdx.x << 3) + (threadIdx.x >> 5);
  if (wavei >= TILESP) return;
  const int n    = blockIdx.y;
  const int lane = threadIdx.x & 31;
  const int m    = lane & 15;
  const int pixBase = wavei << 4;
  const int py   = pixBase / W;               // wave-uniform; strip may span two rows when W % 16 != 0
  const int px   = (pixBase - py*W) + m;
  const int hiA  = (lane & 16) >> 1;          // ISA 16-bit A 16x32: upper half covers K+8
  const int hiB  = (lane & 16) >> 1;          // ISA 16-bit B 32x16: upper half covers K+8
  const int cobase = (int)blockIdx.z * (16*NCT) + m;

  const _Float16* wrow = wgtp + cobase*KTP + hiB; // rows zero-padded to CoutPad16
  #pragma unroll
  for (int j = 0; j < NCT; ++j)
    __builtin_prefetch(wrow + j*16*KTP, 0, 3);    // global_prefetch: warm weight rows

  v8f acc[NCT];
  #pragma unroll
  for (int j = 0; j < NCT; ++j)
    acc[j] = (v8f){0.f,0.f,0.f,0.f,0.f,0.f,0.f,0.f};

  #pragma unroll
  for (int ch = 0; ch < NCHUNK; ++ch) {
    v8h a0 = {}, a1 = {};
    {
      int kb = ch*32 + hiA;
      int t  = kb / CIN;                      // const divisor
      int c  = kb - t*CIN;
      if (t < KK) {
        int iy = py + t / KH - PAD, ix = px + t % KH - PAD;
        if ((unsigned)iy < (unsigned)H && (unsigned)ix < (unsigned)W) {
          if (C1 == CIN || c < C1) {
            int sy = UP ? (iy >> 1) : iy, sx = UP ? (ix >> 1) : ix;
            a0 = *(const v8h*)(inA + ((n*HS + sy)*WS + sx)*C1 + c);
          } else {
            a0 = *(const v8h*)(inB + ((n*H + iy)*W + ix)*C2 + (c - C1));
          }
        }
      }
    }
    {
      int kb = ch*32 + hiA + 16;
      int t  = kb / CIN;
      int c  = kb - t*CIN;
      if (t < KK) {
        int iy = py + t / KH - PAD, ix = px + t % KH - PAD;
        if ((unsigned)iy < (unsigned)H && (unsigned)ix < (unsigned)W) {
          if (C1 == CIN || c < C1) {
            int sy = UP ? (iy >> 1) : iy, sx = UP ? (ix >> 1) : ix;
            a1 = *(const v8h*)(inA + ((n*HS + sy)*WS + sx)*C1 + c);
          } else {
            a1 = *(const v8h*)(inB + ((n*H + iy)*W + ix)*C2 + (c - C1));
          }
        }
      }
    }
    v16h av = __builtin_shufflevector(a0, a1, 0,1,2,3,4,5,6,7,8,9,10,11,12,13,14,15);
    #pragma unroll
    for (int j = 0; j < NCT; ++j) {
      v8h b0 = *(const v8h*)(wrow + j*16*KTP + ch*32);
      v8h b1 = *(const v8h*)(wrow + j*16*KTP + ch*32 + 16);
      v16h bv = __builtin_shufflevector(b0, b1, 0,1,2,3,4,5,6,7,8,9,10,11,12,13,14,15);
      acc[j] = __builtin_amdgcn_wmma_f32_16x16x32_f16(false, av, false, bv, (short)0,
                                                      acc[j], false, false);
    }
  }

  #pragma unroll
  for (int j = 0; j < NCT; ++j) {
    int co = (int)blockIdx.z * (16*NCT) + j*16 + m;
    if (COUT >= 16*NCT || co < COUT) {        // COUT=8 is the only padded case
      float bs = bias[co];
      #pragma unroll
      for (int r = 0; r < 8; ++r) {
        int prow = pixBase + r + ((lane & 16) ? 8 : 0);   // D: lane&15=N, VGPR r -> M
        float vo = acc[j][r] + bs;
        if (RELU) vo = fmaxf(vo, 0.f);
        out[(n*HW + prow)*COUT + co] = (_Float16)vo;
      }
    }
  }
}

// ---------------------------------------------------------------- pool / epilogue (NHWC)
__global__ __launch_bounds__(256) void avgpool_kernel(const _Float16* __restrict__ in,
                                                      _Float16* __restrict__ out,
                                                      int C, int H2, int W2, int total) {
  int idx = blockIdx.x * blockDim.x + threadIdx.x;
  if (idx >= total) return;
  int c  = idx % C;  int t  = idx / C;
  int x2 = t % W2;   t /= W2;
  int y2 = t % H2;   int n = t / H2;
  int W = W2*2;
  const _Float16* b = in + ((n*(H2*2) + y2*2)*W + x2*2)*C + c;
  float s = (float)b[0] + (float)b[C] + (float)b[W*C] + (float)b[W*C + C];
  out[idx] = (_Float16)(0.25f*s);
}

__global__ __launch_bounds__(256) void sweights_kernel(const _Float16* __restrict__ f,
                                                       float* __restrict__ o) {
  int idx = blockIdx.x * blockDim.x + threadIdx.x;
  if (idx >= NVIEW*DNUM*HWFULL) return;
  int p = idx % HWFULL; int t = idx / HWFULL; int d = t % DNUM; int v = t / DNUM;
  o[idx] = (float)f[((d*NVIEW+v)*HWFULL + p)*8 + 0];
}

__global__ __launch_bounds__(256) void gmean_kernel(const _Float16* __restrict__ f,
                                                    _Float16* __restrict__ g) {
  int idx = blockIdx.x * blockDim.x + threadIdx.x;
  if (idx >= DNUM*HWFULL) return;
  int p = idx % HWFULL, d = idx / HWFULL;
  v8h f0 = *(const v8h*)(f + ((d*NVIEW+0)*HWFULL + p)*8);
  v8h f1 = *(const v8h*)(f + ((d*NVIEW+1)*HWFULL + p)*8);
  v8h f2 = *(const v8h*)(f + ((d*NVIEW+2)*HWFULL + p)*8);
  v8h f3 = *(const v8h*)(f + ((d*NVIEW+3)*HWFULL + p)*8);
  v8h o = {};
  #pragma unroll
  for (int c=0;c<7;++c)
    o[c] = (_Float16)(0.25f*((float)f0[1+c]+(float)f1[1+c]+(float)f2[1+c]+(float)f3[1+c]));
  *(v8h*)(g + (d*HWFULL + p)*8) = o;
}

__global__ __launch_bounds__(256) void cell1_kernel(const _Float16* __restrict__ g,
                                                    const float* __restrict__ w,
                                                    const float* __restrict__ b,
                                                    float* __restrict__ o) {
  int idx = blockIdx.x * blockDim.x + threadIdx.x;
  if (idx >= DNUM*HWFULL) return;
  int p = idx % HWFULL, d = idx / HWFULL;
  int y = p / IW, x = p % IW;
  float acc = b[0];
  for (int ky=0;ky<5;++ky){
    int iy = y + ky - 2;
    if ((unsigned)iy >= (unsigned)IH) continue;
    for (int kx=0;kx<5;++kx){
      int ix = x + kx - 2;
      if ((unsigned)ix >= (unsigned)IW) continue;
      v8h gv = *(const v8h*)(g + (d*HWFULL + iy*IW + ix)*8);
      #pragma unroll
      for (int c=0;c<7;++c)
        acc += (float)gv[c] * w[(c*5+ky)*5 + kx];
    }
  }
  o[idx] = acc;
}

// ---------------------------------------------------------------------------- host driver
extern "C" void kernel_launch(void* const* d_in, const int* in_sizes, int n_in,
                              void* d_out, int out_size, void* d_ws, size_t ws_size,
                              hipStream_t stream) {
  (void)out_size; (void)ws_size;
  if (n_in < 29) return;

  const float* src_feats = (const float*)d_in[0];
  const float* dstK = (const float*)d_in[1];
  const float* dstE = (const float*)d_in[2];
  const float* srcK = (const float*)d_in[3];
  const float* srcE = (const float*)d_in[4];

  // Params may arrive in JAX-sorted-key order (cell1_b first, size 1) or dict-insertion order.
  const bool sorted = (in_sizes[5] == 1);
  auto gp = [&](int si, int ii)->const float* { return (const float*)d_in[sorted ? si : ii]; };
  const float* cell1_b = gp(5, 28);  const float* cell1_w = gp(6, 27);
  const float* dec0_b  = gp(7, 22);  const float* dec0_w  = gp(8, 21);
  const float* dec1_b  = gp(9, 24);  const float* dec1_w  = gp(10, 23);
  const float* dec2_b  = gp(11, 26); const float* dec2_w  = gp(12, 25);
  const float* enc0_b  = gp(13, 6);  const float* enc0_w  = gp(14, 5);
  const float* enc1_b  = gp(15, 8);  const float* enc1_w  = gp(16, 7);
  const float* enc2_b  = gp(17, 10); const float* enc2_w  = gp(18, 9);
  const float* enc3_b  = gp(19, 12); const float* enc3_w  = gp(20, 11);
  const float* tr0_b   = gp(21, 14); const float* tr0_w   = gp(22, 13);
  const float* tr1_b   = gp(23, 16); const float* tr1_w   = gp(24, 15);
  const float* tr2_b   = gp(25, 18); const float* tr2_w   = gp(26, 17);
  const float* tr3_b   = gp(27, 20); const float* tr3_w   = gp(28, 19);

  // ---- workspace (all activations NHWC f16, channel counts x8)
  char* ws = (char*)d_ws;
  size_t off = 0;
  auto alloc = [&](size_t bytes)->void* {
    void* p = ws + off; off += (bytes + 255) & ~(size_t)255; return p;
  };
  const size_t HB = sizeof(_Float16);
  const int HW1 = HWFULL, HW2 = HWFULL/4, HW4 = HWFULL/16, HW8 = HWFULL/64;

  _Float16* wc   = (_Float16*)alloc((size_t)64*HW1*WCC*HB);  // 104.9 MB
  _Float16* h0   = (_Float16*)alloc((size_t)64*HW1*8 *HB);
  _Float16* out0 = (_Float16*)alloc((size_t)64*HW1*8 *HB);
  _Float16* pl1  = (_Float16*)alloc((size_t)64*HW2*8 *HB);
  _Float16* h1   = (_Float16*)alloc((size_t)64*HW2*16*HB);
  _Float16* out1 = (_Float16*)alloc((size_t)64*HW2*16*HB);
  _Float16* pl2  = (_Float16*)alloc((size_t)64*HW4*16*HB);
  _Float16* h2   = (_Float16*)alloc((size_t)64*HW4*32*HB);
  _Float16* out2 = (_Float16*)alloc((size_t)64*HW4*32*HB);
  _Float16* pl3  = (_Float16*)alloc((size_t)64*HW8*32*HB);
  _Float16* h3   = (_Float16*)alloc((size_t)64*HW8*64*HB);
  _Float16* out3 = (_Float16*)alloc((size_t)64*HW8*64*HB);
  _Float16* dd0  = (_Float16*)alloc((size_t)64*HW4*32*HB);
  _Float16* dd1  = (_Float16*)alloc((size_t)64*HW2*16*HB);
  float*    mats = (float*)   alloc(64*sizeof(float));
  // wc region is dead after enc0: reuse for f (21 MB @0) and g (5.2 MB @ +48 MB)
  _Float16* fbuf = wc;
  _Float16* gbuf = (_Float16*)((char*)wc + ((size_t)48 << 20));

  auto nblk = [](long n, int bs){ return (unsigned)((n + bs - 1) / bs); };
  auto a16  = [](int x){ return (x + 15) & ~15; };
  auto a32  = [](int x){ return (x + 31) & ~31; };

  // repacked weight buffers: [CoutPad16][KtotPad32] f16
  auto mkw = [&](const float* w, int Cout, int C, int Cpad, int KK)->_Float16* {
    int ktp = a32(KK*Cpad);
    int total = a16(Cout)*ktp;
    _Float16* p = (_Float16*)alloc((size_t)total*HB);
    repack_w_kernel<<<nblk(total,256),256,0,stream>>>(w, p, Cout, C, Cpad, KK, ktp, total);
    return p;
  };
  _Float16* wEnc0 = mkw(enc0_w,  8, 34, WCC, 9);
  _Float16* wEnc1 = mkw(enc1_w, 16,  8,   8, 9);
  _Float16* wEnc2 = mkw(enc2_w, 32, 16,  16, 9);
  _Float16* wEnc3 = mkw(enc3_w, 64, 32,  32, 9);
  _Float16* wTr0  = mkw(tr0_w,   8,  8,   8, 1);
  _Float16* wTr1  = mkw(tr1_w,  16, 16,  16, 1);
  _Float16* wTr2  = mkw(tr2_w,  32, 32,  32, 1);
  _Float16* wTr3  = mkw(tr3_w,  64, 64,  64, 1);
  _Float16* wDec0 = mkw(dec0_w, 32, 96,  96, 9);
  _Float16* wDec1 = mkw(dec1_w, 16, 48,  48, 9);
  _Float16* wDec2 = mkw(dec2_w,  8, 24,  24, 9);

  setup_mats_kernel<<<1, 32, 0, stream>>>(dstK, dstE, srcK, srcE, mats);
  warp_kernel<<<nblk((long)DNUM*NVIEW*HW1, 256), 256, 0, stream>>>(src_feats, mats, wc);
  cost_kernel<<<nblk((long)DNUM*HW1, 256), 256, 0, stream>>>(wc);

  auto grd = [&](int HW, int Cout, int nct){
    int z = Cout / (16*nct); if (z < 1) z = 1;
    return dim3((HW/16 + 7)/8, 64, z);
  };

  // encoder (NCT = cout tiles per wave: A loaded once, reused across accumulators)
  conv_wmma_kernel<3,128,160, WCC, WCC,  8, 1, 0, 1><<<grd(HW1,  8,1), 256, 0, stream>>>(wc,   nullptr, wEnc0, enc0_b, h0);
  conv_wmma_kernel<1,128,160,   8,   8,  8, 0, 0, 1><<<grd(HW1,  8,1), 256, 0, stream>>>(h0,   nullptr, wTr0,  tr0_b,  out0);
  avgpool_kernel<<<nblk(64L*HW2*8, 256),256,0,stream>>>(h0, pl1, 8, 64, 80, 64*HW2*8);
  conv_wmma_kernel<3, 64, 80,   8,   8, 16, 1, 0, 1><<<grd(HW2, 16,1), 256, 0, stream>>>(pl1,  nullptr, wEnc1, enc1_b, h1);
  conv_wmma_kernel<1, 64, 80,  16,  16, 16, 0, 0, 1><<<grd(HW2, 16,1), 256, 0, stream>>>(h1,   nullptr, wTr1,  tr1_b,  out1);
  avgpool_kernel<<<nblk(64L*HW4*16,256),256,0,stream>>>(h1, pl2, 16, 32, 40, 64*HW4*16);
  conv_wmma_kernel<3, 32, 40,  16,  16, 32, 1, 0, 2><<<grd(HW4, 32,2), 256, 0, stream>>>(pl2,  nullptr, wEnc2, enc2_b, h2);
  conv_wmma_kernel<1, 32, 40,  32,  32, 32, 0, 0, 2><<<grd(HW4, 32,2), 256, 0, stream>>>(h2,   nullptr, wTr2,  tr2_b,  out2);
  avgpool_kernel<<<nblk(64L*HW8*32,256),256,0,stream>>>(h2, pl3, 32, 16, 20, 64*HW8*32);
  conv_wmma_kernel<3, 16, 20,  32,  32, 64, 1, 0, 4><<<grd(HW8, 64,4), 256, 0, stream>>>(pl3,  nullptr, wEnc3, enc3_b, h3);
  conv_wmma_kernel<1, 16, 20,  64,  64, 64, 0, 0, 4><<<grd(HW8, 64,4), 256, 0, stream>>>(h3,   nullptr, wTr3,  tr3_b,  out3);

  // decoder: upsample+concat folded into conv input fetch
  conv_wmma_kernel<3, 32, 40,  96,  64, 32, 1, 1, 2><<<grd(HW4, 32,2), 256, 0, stream>>>(out3, out2, wDec0, dec0_b, dd0);
  conv_wmma_kernel<3, 64, 80,  48,  32, 16, 1, 1, 1><<<grd(HW2, 16,1), 256, 0, stream>>>(dd0,  out1, wDec1, dec1_b, dd1);
  conv_wmma_kernel<3,128,160,  24,  16,  8, 1, 1, 1><<<grd(HW1,  8,1), 256, 0, stream>>>(dd1,  out0, wDec2, dec2_b, fbuf);

  // outputs: probs first (16*H*W), then src_weights (4*16*H*W)
  float* probs = (float*)d_out;
  float* swts  = probs + (size_t)DNUM*HW1;
  sweights_kernel<<<nblk((long)NVIEW*DNUM*HW1, 256), 256, 0, stream>>>(fbuf, swts);
  gmean_kernel<<<nblk((long)DNUM*HW1, 256), 256, 0, stream>>>(fbuf, gbuf);
  cell1_kernel<<<nblk((long)DNUM*HW1, 256), 256, 0, stream>>>(gbuf, cell1_w, cell1_b, probs);
}
